// MeshEncoderGCN_79345225826543
// MI455X (gfx1250) — compile-verified, matrix operand order checked by reading, not run on hardware
//
#include <hip/hip_runtime.h>
#include <hip/hip_bf16.h>

// MeshEncoderGCN for MI455X (gfx1250).
// - GEMMs via v_wmma_f32_16x16x32_bf16, 2-term bf16 split (~fp32 accuracy).
//   bf16 hi/lo packing is hoisted out of the GEMM (fused into ReLU / done once
//   per layer for weights), so the WMMA loop is loads + wmma only.
// - Edge aggregation via fp32 global atomics (activations are L2-resident:
//   100K x 300 f32 = 120MB < 192MB L2).
// - Final column-max via int-bitcast atomicMax (valid post-ReLU).

typedef __attribute__((ext_vector_type(16))) __bf16 v16bf;
typedef __attribute__((ext_vector_type(8)))  __bf16 v8bf;
typedef __attribute__((ext_vector_type(8)))  float  v8f;

static __device__ __forceinline__ void bf16_split(float v, __bf16& hi, __bf16& lo) {
  hi = (__bf16)v;
  lo = (__bf16)(v - (float)hi);
}

// ---------------- packing kernels ----------------

// positions (or any f32 x) -> xhi/xlo, K zero-padded to din_pad. One block per row.
__global__ void __launch_bounds__(128)
gcn_pack_x(const float* __restrict__ x, __bf16* __restrict__ xhi, __bf16* __restrict__ xlo,
           int din, int din_pad)
{
  const size_t m = blockIdx.x;
  const float* xr = x + m * din;
  __bf16* ph = xhi + m * din_pad;
  __bf16* pl = xlo + m * din_pad;
  for (int k = threadIdx.x; k < din_pad; k += 128) {
    float v = (k < din) ? xr[k] : 0.f;
    __bf16 h, l; bf16_split(v, h, l);
    ph[k] = h; pl[k] = l;
  }
}

// W (din x dout, row-major) -> transposed+padded Wt_hi/Wt_lo (dout_pad x din_pad).
__global__ void __launch_bounds__(128)
gcn_pack_wt(const float* __restrict__ W, __bf16* __restrict__ whi, __bf16* __restrict__ wlo,
            int din, int dout, int din_pad)
{
  const int n = blockIdx.x;               // padded column index (row of Wt)
  __bf16* ph = whi + (size_t)n * din_pad;
  __bf16* pl = wlo + (size_t)n * din_pad;
  for (int k = threadIdx.x; k < din_pad; k += 128) {
    float v = (n < dout && k < din) ? W[(size_t)k * dout + n] : 0.f;
    __bf16 h, l; bf16_split(v, h, l);
    ph[k] = h; pl[k] = l;
  }
}

// ---------------- WMMA GEMM ----------------

// out = x @ W + bias for (W0,b0)->y and (W1,b1)->h (blockIdx.z selects).
// Block: 256 threads = 8 waves; each wave computes one 16x16 f32 tile.
// Block tile: 16 (M) x 128 (N); K-loop over din_pad in steps of 32.
// Fragment layout (16-bit A 16x32 / mirrored B):
//   k(e,lane) = (e<8 ? 0 : 16) + (lane>=16 ? 8 : 0) + (e&7)
// -> per lane: two contiguous 8-elem (16B) runs => 2 x b128 loads per fragment.
__global__ void __launch_bounds__(256)
gcn_gemm_dual(const __bf16* __restrict__ xhi, const __bf16* __restrict__ xlo,
              const __bf16* __restrict__ w0hi, const __bf16* __restrict__ w0lo,
              const float* __restrict__ b0,
              const __bf16* __restrict__ w1hi, const __bf16* __restrict__ w1lo,
              const float* __restrict__ b1,
              float* __restrict__ y, float* __restrict__ hbuf,
              int N, int din_pad, int dout)
{
  const __bf16* whi = blockIdx.z ? w1hi : w0hi;
  const __bf16* wlo = blockIdx.z ? w1lo : w0lo;
  const float*  bia = blockIdx.z ? b1 : b0;
  float*        out = blockIdx.z ? hbuf : y;

  const int lane = threadIdx.x & 31;
  const int wave = threadIdx.x >> 5;
  const int row  = lane & 15;
  const int koff = (lane >> 4) * 8;
  const int m0   = blockIdx.x * 16;
  int mrow = m0 + row; if (mrow >= N) mrow = N - 1;       // safe clamp (stores guarded)
  const int n = blockIdx.y * 128 + wave * 16 + row;       // < dout_pad, always valid in Wt

  const __bf16* axh = xhi + (size_t)mrow * din_pad + koff;
  const __bf16* axl = xlo + (size_t)mrow * din_pad + koff;
  const __bf16* bwh = whi + (size_t)n    * din_pad + koff;
  const __bf16* bwl = wlo + (size_t)n    * din_pad + koff;

  v8f acc = {};

  for (int k0 = 0; k0 < din_pad; k0 += 32) {
    v8bf ah0 = *(const v8bf*)(axh + k0);
    v8bf ah1 = *(const v8bf*)(axh + k0 + 16);
    v8bf al0 = *(const v8bf*)(axl + k0);
    v8bf al1 = *(const v8bf*)(axl + k0 + 16);
    v8bf bh0 = *(const v8bf*)(bwh + k0);
    v8bf bh1 = *(const v8bf*)(bwh + k0 + 16);
    v8bf bl0 = *(const v8bf*)(bwl + k0);
    v8bf bl1 = *(const v8bf*)(bwl + k0 + 16);

    if (k0 + 32 < din_pad) {   // global_prefetch_b8 of next K-slab
      __builtin_prefetch(axh + k0 + 32, 0, 3);
      __builtin_prefetch(bwh + k0 + 32, 0, 3);
    }

    v16bf ahi = __builtin_shufflevector(ah0, ah1, 0,1,2,3,4,5,6,7,8,9,10,11,12,13,14,15);
    v16bf alo = __builtin_shufflevector(al0, al1, 0,1,2,3,4,5,6,7,8,9,10,11,12,13,14,15);
    v16bf bhi = __builtin_shufflevector(bh0, bh1, 0,1,2,3,4,5,6,7,8,9,10,11,12,13,14,15);
    v16bf blo = __builtin_shufflevector(bl0, bl1, 0,1,2,3,4,5,6,7,8,9,10,11,12,13,14,15);

    // fp32 via bf16 split: hi*hi + hi*lo + lo*hi (lo*lo below noise floor).
    acc = __builtin_amdgcn_wmma_f32_16x16x32_bf16(false, ahi, false, bhi,
                                                  (short)0, acc, false, false);
    acc = __builtin_amdgcn_wmma_f32_16x16x32_bf16(false, ahi, false, blo,
                                                  (short)0, acc, false, false);
    acc = __builtin_amdgcn_wmma_f32_16x16x32_bf16(false, alo, false, bhi,
                                                  (short)0, acc, false, false);
  }

  // C/D layout: VGPR r -> M = r + (lane>=16 ? 8 : 0), N = lane&15.
  float bv = (n < dout) ? bia[n] : 0.f;
  const int mb = m0 + ((lane >> 4) << 3);
#pragma unroll
  for (int r = 0; r < 8; ++r) {
    int m = mb + r;
    if (m < N && n < dout) out[(size_t)m * dout + n] = acc[r] + bv;
  }
}

// ---------------- aggregation / activation / reduction ----------------

// y[src] += h[dst]; y[dst] += h[src]  (one block per edge, coalesced rows)
__global__ void __launch_bounds__(128)
gcn_scatter(const int* __restrict__ edges, const float* __restrict__ h,
            float* __restrict__ y, int dout)
{
  int e = blockIdx.x;
  int s = edges[2 * e + 0];
  int d = edges[2 * e + 1];
  const float* hs = h + (size_t)s * dout;
  const float* hd = h + (size_t)d * dout;
  float* ys = y + (size_t)s * dout;
  float* yd = y + (size_t)d * dout;
  for (int f = threadIdx.x; f < dout; f += 128) {
    atomicAdd(&ys[f], hd[f]);
    atomicAdd(&yd[f], hs[f]);
  }
}

// ReLU in place + pack next layer's bf16 hi/lo activations. One block per row.
__global__ void __launch_bounds__(128)
gcn_relu_pack(float* __restrict__ y, __bf16* __restrict__ xhi, __bf16* __restrict__ xlo,
              int dout, int dpad_next)
{
  const size_t m = blockIdx.x;
  float* yr = y + m * dout;
  __bf16* ph = xhi + m * dpad_next;
  __bf16* pl = xlo + m * dpad_next;
  for (int k = threadIdx.x; k < dpad_next; k += 128) {
    float v = 0.f;
    if (k < dout) { v = fmaxf(yr[k], 0.f); yr[k] = v; }
    __bf16 h, l; bf16_split(v, h, l);
    ph[k] = h; pl[k] = l;
  }
}

__global__ void __launch_bounds__(256)
gcn_relu(float* __restrict__ y, long long n)
{
  long long i = (long long)blockIdx.x * blockDim.x + threadIdx.x;
  long long stride = (long long)gridDim.x * blockDim.x;
  for (; i < n; i += stride) y[i] = fmaxf(y[i], 0.f);
}

__global__ void gcn_zero_out(int* __restrict__ out, int n)
{
  int i = threadIdx.x;
  if (i < n) out[i] = 0;  // 0.0f bit pattern; valid lower bound post-ReLU
}

// Column max over N rows; post-ReLU values are >= 0 so float bits order as ints.
__global__ void __launch_bounds__(128)
gcn_colmax(const float* __restrict__ x, int* __restrict__ out, int N, int dout)
{
  int f = threadIdx.x;
  if (f >= dout) return;
  float m = 0.f;
  for (int r = blockIdx.x; r < N; r += gridDim.x)
    m = fmaxf(m, x[(size_t)r * dout + f]);
  atomicMax(&out[f], __float_as_int(m));
}

// ---------------- host orchestration ----------------

extern "C" void kernel_launch(void* const* d_in, const int* in_sizes, int n_in,
                              void* d_out, int out_size, void* d_ws, size_t ws_size,
                              hipStream_t stream)
{
  static const int DIMS[18] = {3, 60, 60, 60, 60, 120, 120, 120, 150,
                               200, 210, 250, 300, 300, 300, 300, 300, 128};
  const int NL = 17;
  const int N = in_sizes[0] / 3;   // 100000 nodes
  const int E = in_sizes[1] / 2;   // 300000 edges
  const int* edges = (const int*)d_in[1];
  (void)ws_size; (void)n_in;

  const int KPAD_MAX = 320;   // max din rounded up to 32
  const int NPAD_MAX = 384;   // max dout rounded up to 128

  // Workspace arena (256B-aligned slices).
  char* p = (char*)d_ws;
  auto take = [&](size_t bytes) {
    char* r = p;
    p += (bytes + 255) & ~(size_t)255;
    return r;
  };
  float*  y    = (float*)take((size_t)N * 300 * sizeof(float));
  float*  h    = (float*)take((size_t)N * 300 * sizeof(float));
  __bf16* xhi  = (__bf16*)take((size_t)N * KPAD_MAX * 2);
  __bf16* xlo  = (__bf16*)take((size_t)N * KPAD_MAX * 2);
  __bf16* w0hi = (__bf16*)take((size_t)NPAD_MAX * KPAD_MAX * 2);
  __bf16* w0lo = (__bf16*)take((size_t)NPAD_MAX * KPAD_MAX * 2);
  __bf16* w1hi = (__bf16*)take((size_t)NPAD_MAX * KPAD_MAX * 2);
  __bf16* w1lo = (__bf16*)take((size_t)NPAD_MAX * KPAD_MAX * 2);

  // Pack the input positions (layer 0 activations).
  {
    int din = DIMS[0], din_pad = (din + 31) & ~31;
    gcn_pack_x<<<N, 128, 0, stream>>>((const float*)d_in[0], xhi, xlo, din, din_pad);
  }

  for (int i = 0; i < NL; ++i) {
    const int din      = DIMS[i];
    const int dout     = DIMS[i + 1];
    const int din_pad  = (din + 31) & ~31;
    const int dout_pad = ((dout + 127) / 128) * 128;
    const float* W0 = (const float*)d_in[2 + 4 * i + 0];
    const float* b0 = (const float*)d_in[2 + 4 * i + 1];
    const float* W1 = (const float*)d_in[2 + 4 * i + 2];
    const float* b1 = (const float*)d_in[2 + 4 * i + 3];

    gcn_pack_wt<<<dout_pad, 128, 0, stream>>>(W0, w0hi, w0lo, din, dout, din_pad);
    gcn_pack_wt<<<dout_pad, 128, 0, stream>>>(W1, w1hi, w1lo, din, dout, din_pad);

    dim3 grid((N + 15) / 16, dout_pad / 128, 2);
    gcn_gemm_dual<<<grid, 256, 0, stream>>>(xhi, xlo, w0hi, w0lo, b0,
                                            w1hi, w1lo, b1, y, h, N, din_pad, dout);

    gcn_scatter<<<E, 128, 0, stream>>>(edges, h, y, dout);

    if (i + 1 < NL) {
      int dpad_next = (dout + 31) & ~31;
      gcn_relu_pack<<<N, 128, 0, stream>>>(y, xhi, xlo, dout, dpad_next);
    } else {
      gcn_relu<<<4096, 256, 0, stream>>>(y, (long long)N * dout);
    }
  }

  gcn_zero_out<<<1, 128, 0, stream>>>((int*)d_out, out_size);
  gcn_colmax<<<512, 128, 0, stream>>>(y, (int*)d_out, N, DIMS[NL]);
}
